// FinancialSentimentLoRA_25555055411403
// MI455X (gfx1250) — matrix operand order repaired and checked by reading, not compiled, not run on hardware
//
#include <hip/hip_runtime.h>
#include <hip/hip_bf16.h>

// ---------------------------------------------------------------------------
// Problem constants (from reference)
// ---------------------------------------------------------------------------
#define Bc    8
#define Sc    1024
#define Dc    1024
#define Hc    16
#define HDc   64
#define Rc    8
#define DFFc  4096
#define DCc   512
#define NCc   3
#define Mrows (Bc*Sc)          // 8192
#define SCALINGc 2.0f          // 16.0 / R

typedef __bf16 bf16t;
typedef __bf16  v16bf __attribute__((ext_vector_type(16)));
typedef float   v8f   __attribute__((ext_vector_type(8)));
typedef int     v4i_t __attribute__((ext_vector_type(4)));

#define WMMA_BF16(a,b,c) \
  __builtin_amdgcn_wmma_f32_16x16x32_bf16(false,(a),false,(b),(short)0,(c),false,false)

union ABfrag { v16bf v; uint4 q[2]; };

// ---------------------------------------------------------------------------
// CDNA5 async global->LDS copy (ASYNCcnt-tracked DMA into LDS, no dest VGPRs).
// Builtin prototype wants v4i addrspace(1)* / v4i addrspace(3)* pointers.
// ---------------------------------------------------------------------------
#if defined(__has_builtin)
#if __has_builtin(__builtin_amdgcn_global_load_async_to_lds_b128)
#define HAVE_ASYNC_LDS 1
#endif
#endif
#ifndef HAVE_ASYNC_LDS
#define HAVE_ASYNC_LDS 0
#endif

#define AS1Q __attribute__((address_space(1)))
#define AS3Q __attribute__((address_space(3)))

__device__ inline void async_copy16(const bf16t* g, bf16t* l){
#if HAVE_ASYNC_LDS
  __builtin_amdgcn_global_load_async_to_lds_b128(
      (AS1Q v4i_t*)(v4i_t*)g,
      (AS3Q v4i_t*)(v4i_t*)l,
      0, 0);
#else
  *(uint4*)l = *(const uint4*)g;
#endif
}

__device__ inline void async_wait0(){
#if HAVE_ASYNC_LDS
#if defined(__has_builtin) && __has_builtin(__builtin_amdgcn_s_wait_asynccnt)
  __builtin_amdgcn_s_wait_asynccnt(0);
#else
  asm volatile("s_wait_asynccnt 0x0" ::: "memory");
#endif
#endif
}

__device__ inline v8f v8f_zero(){ v8f z;
#pragma unroll
  for (int i=0;i<8;i++) z[i]=0.0f; return z; }

__device__ inline float geluf(float x){
  return 0.5f*x*(1.0f + erff(x*0.70710678118654752f));
}

// ---------------------------------------------------------------------------
// fp32 -> bf16 conversion
// ---------------------------------------------------------------------------
__global__ void k_f32_to_bf16(const float* __restrict__ in, bf16t* __restrict__ out, int n){
  int i = blockIdx.x*blockDim.x + threadIdx.x;
  if (i < n) out[i] = (bf16t)in[i];
}

// ---------------------------------------------------------------------------
// LayerNorm: fp32 [row, Dc] -> bf16 normalized; one block (256 thr) per row
// ---------------------------------------------------------------------------
__global__ __launch_bounds__(256)
void k_layernorm_bf16(const float* __restrict__ x, const float* __restrict__ g,
                      const float* __restrict__ beta, bf16t* __restrict__ out){
  __shared__ float red[256];
  int row = blockIdx.x;
  const float* xr = x + (size_t)row*Dc;
  float v[4], s1=0.f, s2=0.f;
#pragma unroll
  for (int i=0;i<4;i++){ v[i]=xr[threadIdx.x + i*256]; s1+=v[i]; s2+=v[i]*v[i]; }
  red[threadIdx.x]=s1; __syncthreads();
  for (int st=128; st>0; st>>=1){ if (threadIdx.x<st) red[threadIdx.x]+=red[threadIdx.x+st]; __syncthreads(); }
  float mu = red[0]*(1.0f/Dc); __syncthreads();
  red[threadIdx.x]=s2; __syncthreads();
  for (int st=128; st>0; st>>=1){ if (threadIdx.x<st) red[threadIdx.x]+=red[threadIdx.x+st]; __syncthreads(); }
  float var = red[0]*(1.0f/Dc) - mu*mu;
  float rstd = rsqrtf(var + 1e-5f);
#pragma unroll
  for (int i=0;i<4;i++){
    int d = threadIdx.x + i*256;
    out[(size_t)row*Dc + d] = (bf16t)((v[i]-mu)*rstd*g[d] + beta[d]);
  }
}

// ---------------------------------------------------------------------------
// LoRA down-projection t[M,8] = h(bf16) @ A^T  (A fp32 [8, Dc])
// one wave per row, 8 waves per block
// ---------------------------------------------------------------------------
__global__ __launch_bounds__(256)
void k_lora_t(const bf16t* __restrict__ h, const float* __restrict__ A,
              float* __restrict__ t){
  int wave = threadIdx.x >> 5, lane = threadIdx.x & 31;
  int m = blockIdx.x*8 + wave;
  const bf16t* hr = h + (size_t)m*Dc;
  float acc[Rc];
#pragma unroll
  for (int r=0;r<Rc;r++) acc[r]=0.f;
  for (int d=lane; d<Dc; d+=32){
    float hv = (float)hr[d];
#pragma unroll
    for (int r=0;r<Rc;r++) acc[r] += hv * A[r*Dc + d];
  }
#pragma unroll
  for (int r=0;r<Rc;r++)
    for (int s=1;s<32;s<<=1) acc[r] += __shfl_xor(acc[r], s, 32);
  if (lane==0){
#pragma unroll
    for (int r=0;r<Rc;r++) t[(size_t)m*Rc + r] = acc[r];
  }
}

// ---------------------------------------------------------------------------
// WMMA GEMM:  C[M,N] = A[M,K](bf16) * W[N,K]^T(bf16)  (+ epilogue)
// 128x128 block tile, BK=32, 256 threads = 8 waves (wave32),
// wave tile 32x64 = 2x4 wmma 16x16 accumulators.
// LDS double-buffered; tiles staged with async global->LDS DMA,
// next tile in flight while WMMAs run on the current one.
// Epilogues:
//   1: + SCALING * loraT[row,:8] . loraB[col,:8]            -> bf16 out
//   2: epi 1 + resid[row,col]                               -> f32 out
//   3: + bias[col], GELU                                    -> bf16 out
//   4: + bias[col] + resid[row,col]                         -> f32 out
// ---------------------------------------------------------------------------
#define GLDT 40   // LDS row stride (bf16 elems): 80 B, 16-B aligned rows

template<int EPI>
__global__ __launch_bounds__(256)
void k_gemm(const bf16t* __restrict__ Aw, const bf16t* __restrict__ Ww,
            int M, int N, int K,
            const float* __restrict__ bias,
            const float* __restrict__ loraT, const float* __restrict__ loraB,
            const float* __restrict__ resid,
            float* __restrict__ outF, bf16t* __restrict__ outB){
  __shared__ __align__(16) bf16t As[2][128*GLDT];
  __shared__ __align__(16) bf16t Bs[2][128*GLDT];
  const int m0 = blockIdx.y*128, n0 = blockIdx.x*128;
  const int lane = threadIdx.x & 31, wave = threadIdx.x >> 5;
  const int lh = lane & 15, l16 = lane >> 4;
  const int wm = (wave & 3)*32, wn = (wave >> 2)*64;

  auto stage = [&](int bufi, int kb){
#pragma unroll
    for (int it=0; it<2; ++it){
      int idx = threadIdx.x + it*256;
      int r = idx >> 2, cv = (idx & 3)*8;
      async_copy16(&Aw[(size_t)(m0+r)*K + kb + cv], &As[bufi][r*GLDT + cv]);
      async_copy16(&Ww[(size_t)(n0+r)*K + kb + cv], &Bs[bufi][r*GLDT + cv]);
    }
#if !HAVE_ASYNC_LDS
    if (kb + 32 < K){
      int r = threadIdx.x >> 2, cv = (threadIdx.x & 3)*8;
      __builtin_prefetch(&Aw[(size_t)(m0+r)*K + kb + 32 + cv], 0, 1);
      __builtin_prefetch(&Ww[(size_t)(n0+r)*K + kb + 32 + cv], 0, 1);
    }
#endif
  };

  v8f acc[2][4];
#pragma unroll
  for (int i=0;i<2;i++)
#pragma unroll
    for (int j=0;j<4;j++) acc[i][j] = v8f_zero();

  stage(0, 0);
  int buf = 0;
  for (int kb=0; kb<K; kb+=32){
    async_wait0();
    __syncthreads();
    if (kb + 32 < K) stage(buf^1, kb + 32);

    ABfrag fa[2], fb[4];
#pragma unroll
    for (int i=0;i<2;i++){
      int rr = wm + i*16 + lh;
      int ba = rr*GLDT + l16*8;              // elems 0..7: K = kb2..kb2+7
      fa[i].q[0] = *(const uint4*)&As[buf][ba];
      fa[i].q[1] = *(const uint4*)&As[buf][ba + 16];  // elems 8..15: +16
    }
#pragma unroll
    for (int j=0;j<4;j++){
      int nn = wn + j*16 + lh;
      int bb = nn*GLDT + l16*8;              // B: same K order as A (8h+s, then +16)
      fb[j].q[0] = *(const uint4*)&Bs[buf][bb];
      fb[j].q[1] = *(const uint4*)&Bs[buf][bb + 16];
    }
#pragma unroll
    for (int i=0;i<2;i++)
#pragma unroll
      for (int j=0;j<4;j++)
        acc[i][j] = WMMA_BF16(fa[i].v, fb[j].v, acc[i][j]);
    buf ^= 1;
  }

  // Epilogue (C layout: lanes 0-15 -> M = e, N = lane; lanes 16-31 -> M = 8+e)
  float brv[4][Rc];
  if (EPI==1 || EPI==2){
#pragma unroll
    for (int j=0;j<4;j++){
      int col = n0 + wn + j*16 + lh;
#pragma unroll
      for (int r=0;r<Rc;r++) brv[j][r] = loraB[(size_t)col*Rc + r];
    }
  }
#pragma unroll
  for (int i=0;i<2;i++){
#pragma unroll
    for (int e=0;e<8;e++){
      int row = m0 + wm + i*16 + (l16 ? e+8 : e);
      float trv[Rc];
      if (EPI==1 || EPI==2){
#pragma unroll
        for (int r=0;r<Rc;r++) trv[r] = loraT[(size_t)row*Rc + r];
      }
#pragma unroll
      for (int j=0;j<4;j++){
        int col = n0 + wn + j*16 + lh;
        float v = acc[i][j][e];
        if (EPI==1 || EPI==2){
          float lr = 0.f;
#pragma unroll
          for (int r=0;r<Rc;r++) lr += trv[r]*brv[j][r];
          v += SCALINGc * lr;
        }
        if (EPI==3 || EPI==4) v += bias[col];
        if (EPI==3) v = geluf(v);
        if (EPI==2 || EPI==4) v += resid[(size_t)row*N + col];
        if (EPI==2 || EPI==4) outF[(size_t)row*N + col] = v;
        else                  outB[(size_t)row*N + col] = (bf16t)v;
      }
    }
  }
}

// ---------------------------------------------------------------------------
// Flash attention, one block = 128 query rows of one (b,h); 8 waves,
// each wave owns 16 query rows. K/V tiles of 64 keys streamed through LDS.
// q,k,v,ctx all bf16 [B*S, Dc] with head h at columns h*64..h*64+63.
// ---------------------------------------------------------------------------
#define AQS 72   // LDS row stride (bf16 elems): 144 B, 16-B aligned

__global__ __launch_bounds__(256)
void k_attention(const bf16t* __restrict__ qb, const bf16t* __restrict__ kb,
                 const bf16t* __restrict__ vb, bf16t* __restrict__ ctxb){
  __shared__ __align__(16) bf16t Qs [128*AQS];
  __shared__ __align__(16) bf16t Ks [64*AQS];
  __shared__ __align__(16) bf16t VsT[64*AQS];   // transposed: [hd][pos]
  __shared__ __align__(16) bf16t Ps [128*AQS];

  const int bh = blockIdx.y, b = bh >> 4, h = bh & 15;
  const int q0 = blockIdx.x * 128;
  const int lane = threadIdx.x & 31, wave = threadIdx.x >> 5;
  const int lh = lane & 15, l16 = lane >> 4;
  const size_t rowbase = (size_t)b*Sc;
  const int colbase = h*HDc;

  // Load Q tile 128x64 (async DMA into LDS)
#pragma unroll
  for (int it=0; it<4; ++it){
    int idx = threadIdx.x + it*256;
    int r = idx >> 3, cv = (idx & 7)*8;
    async_copy16(&qb[(rowbase + q0 + r)*Dc + colbase + cv], &Qs[r*AQS + cv]);
  }
  async_wait0();
  __syncthreads();

  ABfrag qa[2];
#pragma unroll
  for (int t=0;t<2;t++){
    int rr = wave*16 + lh;
    int ba = rr*AQS + t*32 + l16*8;
    qa[t].q[0] = *(const uint4*)&Qs[ba];
    qa[t].q[1] = *(const uint4*)&Qs[ba + 16];
  }

  float mrow[8], lsum[8];
  v8f cacc[4];
#pragma unroll
  for (int e=0;e<8;e++){ mrow[e] = -3.0e30f; lsum[e] = 0.f; }
#pragma unroll
  for (int j=0;j<4;j++) cacc[j] = v8f_zero();

  for (int kt=0; kt<Sc; kt+=64){
    __syncthreads();
    // K tile via async DMA; V tile loaded and transposed manually
#pragma unroll
    for (int it=0; it<2; ++it){
      int idx = threadIdx.x + it*256;
      int r = idx >> 3, cv = (idx & 7)*8;
      async_copy16(&kb[(rowbase + kt + r)*Dc + colbase + cv], &Ks[r*AQS + cv]);
      union { uint4 q; bf16t hv[8]; } tv;
      tv.q = *(const uint4*)&vb[(rowbase + kt + r)*Dc + colbase + cv];
#pragma unroll
      for (int u=0;u<8;u++) VsT[(cv+u)*AQS + r] = tv.hv[u];
    }
    async_wait0();
    __syncthreads();

    // scores S = Q K^T * 1/sqrt(64)
    v8f sacc[4];
#pragma unroll
    for (int j=0;j<4;j++){
      sacc[j] = v8f_zero();
#pragma unroll
      for (int t=0;t<2;t++){
        ABfrag bk;
        int nn = j*16 + lh;
        int ba = nn*AQS + t*32 + l16*8;
        bk.q[0] = *(const uint4*)&Ks[ba];
        bk.q[1] = *(const uint4*)&Ks[ba + 16];
        sacc[j] = WMMA_BF16(qa[t].v, bk.v, sacc[j]);
      }
      sacc[j] = sacc[j] * 0.125f;
    }

    // online softmax (per row; rows live on 16-lane halves)
#pragma unroll
    for (int e=0;e<8;e++){
      float ml = fmaxf(fmaxf(sacc[0][e], sacc[1][e]), fmaxf(sacc[2][e], sacc[3][e]));
#pragma unroll
      for (int s=1;s<16;s<<=1) ml = fmaxf(ml, __shfl_xor(ml, s, 32));
      float mn = fmaxf(mrow[e], ml);
      float sc = __expf(mrow[e] - mn);
      mrow[e] = mn;
      lsum[e] *= sc;
#pragma unroll
      for (int j=0;j<4;j++) cacc[j][e] *= sc;
      int pr = wave*16 + (l16 ? e+8 : e);
#pragma unroll
      for (int j=0;j<4;j++){
        float p = __expf(sacc[j][e] - mn);
        lsum[e] += p;
        Ps[pr*AQS + j*16 + lh] = (bf16t)p;
      }
    }
    asm volatile("" ::: "memory");   // keep P stores before P reads (same wave)

    // ctx += P V
    ABfrag pa[2];
#pragma unroll
    for (int t=0;t<2;t++){
      int rr = wave*16 + lh;
      int ba = rr*AQS + t*32 + l16*8;
      pa[t].q[0] = *(const uint4*)&Ps[ba];
      pa[t].q[1] = *(const uint4*)&Ps[ba + 16];
    }
#pragma unroll
    for (int jj=0;jj<4;jj++){
#pragma unroll
      for (int t=0;t<2;t++){
        ABfrag bv;
        int nn = jj*16 + lh;
        int ba = nn*AQS + t*32 + l16*8;
        bv.q[0] = *(const uint4*)&VsT[ba];
        bv.q[1] = *(const uint4*)&VsT[ba + 16];
        cacc[jj] = WMMA_BF16(pa[t].v, bv.v, cacc[jj]);
      }
    }
  }

  // finalize: divide by row sum, store bf16 ctx
#pragma unroll
  for (int e=0;e<8;e++){
    float l = lsum[e];
#pragma unroll
    for (int s=1;s<16;s<<=1) l += __shfl_xor(l, s, 32);
    float inv = 1.0f / l;
    int gr = q0 + wave*16 + (l16 ? e+8 : e);
#pragma unroll
    for (int jj=0;jj<4;jj++)
      ctxb[(rowbase + gr)*Dc + colbase + jj*16 + lh] = (bf16t)(cacc[jj][e]*inv);
  }
}

// ---------------------------------------------------------------------------
// Classifier head: pooled = h2[:,0]; logits = gelu(pooled Wc1^T + bc1) Wc2^T + bc2
// ---------------------------------------------------------------------------
__global__ __launch_bounds__(256)
void k_classifier(const float* __restrict__ h2, const float* __restrict__ Wc1,
                  const float* __restrict__ bc1, const float* __restrict__ Wc2,
                  const float* __restrict__ bc2, float* __restrict__ out){
  __shared__ float pool[Dc];
  __shared__ float c1[DCc];
  int b = blockIdx.x;
  const float* pr = h2 + (size_t)b*Sc*Dc;   // token 0 of batch b
  for (int i=threadIdx.x; i<Dc; i+=256) pool[i] = pr[i];
  __syncthreads();
  for (int n=threadIdx.x; n<DCc; n+=256){
    float a = bc1[n];
    const float* w = Wc1 + (size_t)n*Dc;
    for (int d=0; d<Dc; ++d) a += pool[d]*w[d];
    c1[n] = geluf(a);
  }
  __syncthreads();
  if (threadIdx.x < NCc){
    int n = threadIdx.x;
    float a = bc2[n];
    const float* w = Wc2 + (size_t)n*DCc;
    for (int d=0; d<DCc; ++d) a += c1[d]*w[d];
    out[b*NCc + n] = a;
  }
}

// ---------------------------------------------------------------------------
// Host launch
// ---------------------------------------------------------------------------
extern "C" void kernel_launch(void* const* d_in, const int* in_sizes, int n_in,
                              void* d_out, int out_size, void* d_ws, size_t ws_size,
                              hipStream_t stream) {
  const float* hidden = (const float*)d_in[0];
  const float* Wq = (const float*)d_in[1];
  const float* Aq = (const float*)d_in[2];
  const float* Bq = (const float*)d_in[3];
  const float* Wk = (const float*)d_in[4];
  const float* Ak = (const float*)d_in[5];
  const float* Bk = (const float*)d_in[6];
  const float* Wv = (const float*)d_in[7];
  const float* Av = (const float*)d_in[8];
  const float* Bv = (const float*)d_in[9];
  const float* Wo = (const float*)d_in[10];
  const float* Ao = (const float*)d_in[11];
  const float* Bo = (const float*)d_in[12];
  const float* ln1_g = (const float*)d_in[13];
  const float* ln1_b = (const float*)d_in[14];
  const float* ln2_g = (const float*)d_in[15];
  const float* ln2_b = (const float*)d_in[16];
  const float* W1 = (const float*)d_in[17];
  const float* b1 = (const float*)d_in[18];
  const float* W2 = (const float*)d_in[19];
  const float* b2 = (const float*)d_in[20];
  const float* Wc1 = (const float*)d_in[21];
  const float* bc1 = (const float*)d_in[22];
  const float* Wc2 = (const float*)d_in[23];
  const float* bc2 = (const float*)d_in[24];

  char* wsp = (char*)d_ws;
  auto alloc = [&](size_t bytes)->char*{
    char* p = wsp; wsp += (bytes + 255) & ~(size_t)255; return p;
  };
  bf16t* hnb  = (bf16t*)alloc((size_t)Mrows*Dc*sizeof(bf16t));
  bf16t* Wqb  = (bf16t*)alloc((size_t)Dc*Dc*sizeof(bf16t));
  bf16t* Wkb  = (bf16t*)alloc((size_t)Dc*Dc*sizeof(bf16t));
  bf16t* Wvb  = (bf16t*)alloc((size_t)Dc*Dc*sizeof(bf16t));
  bf16t* Wob  = (bf16t*)alloc((size_t)Dc*Dc*sizeof(bf16t));
  bf16t* W1b  = (bf16t*)alloc((size_t)DFFc*Dc*sizeof(bf16t));
  bf16t* W2b  = (bf16t*)alloc((size_t)Dc*DFFc*sizeof(bf16t));
  bf16t* qbuf = (bf16t*)alloc((size_t)Mrows*Dc*sizeof(bf16t));
  bf16t* kbuf = (bf16t*)alloc((size_t)Mrows*Dc*sizeof(bf16t));
  bf16t* vbuf = (bf16t*)alloc((size_t)Mrows*Dc*sizeof(bf16t));
  bf16t* ctxb = (bf16t*)alloc((size_t)Mrows*Dc*sizeof(bf16t));
  bf16t* hn2b = (bf16t*)alloc((size_t)Mrows*Dc*sizeof(bf16t));
  bf16t* ffb  = (bf16t*)alloc((size_t)Mrows*DFFc*sizeof(bf16t));
  float* h1   = (float*)alloc((size_t)Mrows*Dc*sizeof(float));
  float* h2   = (float*)alloc((size_t)Mrows*Dc*sizeof(float));
  float* tq   = (float*)alloc((size_t)Mrows*Rc*sizeof(float));
  float* tk   = (float*)alloc((size_t)Mrows*Rc*sizeof(float));
  float* tv   = (float*)alloc((size_t)Mrows*Rc*sizeof(float));
  float* to   = (float*)alloc((size_t)Mrows*Rc*sizeof(float));
  (void)ws_size; (void)in_sizes; (void)n_in; (void)out_size;

  auto conv = [&](const float* in, bf16t* out, int n){
    k_f32_to_bf16<<<(n+255)/256, 256, 0, stream>>>(in, out, n);
  };
  conv(Wq, Wqb, Dc*Dc);  conv(Wk, Wkb, Dc*Dc);
  conv(Wv, Wvb, Dc*Dc);  conv(Wo, Wob, Dc*Dc);
  conv(W1, W1b, DFFc*Dc); conv(W2, W2b, Dc*DFFc);

  // LN1 -> bf16
  k_layernorm_bf16<<<Mrows, 256, 0, stream>>>(hidden, ln1_g, ln1_b, hnb);

  // LoRA down-projections
  k_lora_t<<<Mrows/8, 256, 0, stream>>>(hnb, Aq, tq);
  k_lora_t<<<Mrows/8, 256, 0, stream>>>(hnb, Ak, tk);
  k_lora_t<<<Mrows/8, 256, 0, stream>>>(hnb, Av, tv);

  // Q/K/V projections (WMMA) with LoRA epilogue -> bf16
  dim3 gqkv(Dc/128, Mrows/128);
  k_gemm<1><<<gqkv, 256, 0, stream>>>(hnb, Wqb, Mrows, Dc, Dc,
      nullptr, tq, Bq, nullptr, nullptr, qbuf);
  k_gemm<1><<<gqkv, 256, 0, stream>>>(hnb, Wkb, Mrows, Dc, Dc,
      nullptr, tk, Bk, nullptr, nullptr, kbuf);
  k_gemm<1><<<gqkv, 256, 0, stream>>>(hnb, Wvb, Mrows, Dc, Dc,
      nullptr, tv, Bv, nullptr, nullptr, vbuf);

  // Flash attention (WMMA)
  dim3 gattn(Sc/128, Bc*Hc);
  k_attention<<<gattn, 256, 0, stream>>>(qbuf, kbuf, vbuf, ctxb);

  // Output projection + LoRA + residual -> h1 (fp32)
  k_lora_t<<<Mrows/8, 256, 0, stream>>>(ctxb, Ao, to);
  k_gemm<2><<<gqkv, 256, 0, stream>>>(ctxb, Wob, Mrows, Dc, Dc,
      nullptr, to, Bo, hidden, h1, nullptr);

  // LN2 -> bf16
  k_layernorm_bf16<<<Mrows, 256, 0, stream>>>(h1, ln2_g, ln2_b, hn2b);

  // FFN1: gelu(hn2 W1^T + b1) -> bf16
  dim3 gf1(DFFc/128, Mrows/128);
  k_gemm<3><<<gf1, 256, 0, stream>>>(hn2b, W1b, Mrows, DFFc, Dc,
      b1, nullptr, nullptr, nullptr, nullptr, ffb);

  // FFN2: ff W2^T + b2 + h1 -> h2 (fp32)
  dim3 gf2(Dc/128, Mrows/128);
  k_gemm<4><<<gf2, 256, 0, stream>>>(ffb, W2b, Mrows, Dc, DFFc,
      b2, nullptr, nullptr, h1, h2, nullptr);

  // Classifier head
  k_classifier<<<Bc, 256, 0, stream>>>(h2, Wc1, bc1, Wc2, bc2, (float*)d_out);
}
